// GCN_Pool_29300266893895
// MI455X (gfx1250) — compile-verified, matrix operand order checked by reading, not run on hardware
//
#include <hip/hip_runtime.h>
#include <math.h>

#define N_NODES 100000
#define N_EDGES 3200000
#define F_IN    512
#define HID     32
#define NGRAPH  256
#define NTILES  (N_NODES / 16)   // 6250, exact

typedef __attribute__((ext_vector_type(16))) _Float16 v16h;
typedef __attribute__((ext_vector_type(8)))  _Float16 v8h;
typedef __attribute__((ext_vector_type(8)))  float    v8f;

static __device__ __forceinline__ void atomicMaxF(float* addr, float v) {
    int old = __float_as_int(*addr);
    while (__int_as_float(old) < v) {
        int prev = atomicCAS((int*)addr, old, __float_as_int(v));
        if (prev == old) break;
        old = prev;
    }
}

// ---------------- init / degree / counts ----------------
__global__ void k_fill(float* p, float v, long n) {
    long i = (long)blockIdx.x * blockDim.x + threadIdx.x;
    if (i < n) p[i] = v;
}

__global__ void k_deg(const int* __restrict__ dst, float* __restrict__ deg, int e) {
    int i = blockIdx.x * blockDim.x + threadIdx.x;
    if (i < e) atomicAdd(&deg[dst[i]], 1.0f);
}

__global__ void k_cnt(const int* __restrict__ batch, float* __restrict__ cnt, int n) {
    int i = blockIdx.x * blockDim.x + threadIdx.x;
    if (i < n) atomicAdd(&cnt[batch[i]], 1.0f);
}

__global__ void k_dinv(float* d, int n) {
    int i = blockIdx.x * blockDim.x + threadIdx.x;
    if (i < n) d[i] = rsqrtf(d[i] + 1.0f);   // +1 self-loop; deg>=1 always
}

// ---------------- weight fragment prep (f32 -> f16, WMMA-B layout) ----------------
// Fragment: index ((kt*NT + nt)*32 + lane)*16 + j ; lanes 0-15 hold col n=lane with
// K = {0..7, 16..23}+kbase, lanes 16-31 hold col n=lane-16 with K = {8..15, 24..31}+kbase.
__global__ void k_wprep1(const float* __restrict__ W, _Float16* __restrict__ wf) {
    int idx = blockIdx.x * blockDim.x + threadIdx.x;
    if (idx >= 16 * 4 * 32 * 16) return;
    int j = idx & 15, lane = (idx >> 4) & 31, nt = (idx >> 9) & 3, kt = idx >> 11;
    int koff = (j < 8 ? j : j + 8) + ((lane & 16) ? 8 : 0);
    int k = kt * 32 + koff;
    int c = nt * 16 + (lane & 15);
    float v = (c < HID) ? W[k * HID + c] : W[(F_IN + k) * HID + (c - HID)];
    wf[idx] = (_Float16)v;
}

__global__ void k_wprep_s(const float* __restrict__ W, _Float16* __restrict__ wf) {
    int idx = blockIdx.x * blockDim.x + threadIdx.x;
    if (idx >= 2 * 32 * 16) return;
    int j = idx & 15, lane = (idx >> 4) & 31, nt = (idx >> 9) & 1;
    int koff = (j < 8 ? j : j + 8) + ((lane & 16) ? 8 : 0);
    int c = nt * 16 + (lane & 15);
    wf[idx] = (_Float16)W[koff * HID + c];   // rows 0..31 = "a" half of [64,32]
}

// ---------------- layer-1 GEMM: x[N,512] @ Wcat[512,64] via WMMA ----------------
// cols 0..31 -> t (node-local u part); cols 32..63 -> atomic segment-sum sv (v part)
__global__ void k_gemm1(const float* __restrict__ x, const _Float16* __restrict__ wf,
                        const int* __restrict__ batch,
                        float* __restrict__ t, float* __restrict__ sv, int ntiles) {
    __shared__ __align__(32) _Float16 sW[F_IN * 64];   // 64 KB
    for (int i = threadIdx.x; i < (F_IN * 64) / 8; i += blockDim.x)
        ((v8h*)sW)[i] = ((const v8h*)wf)[i];
    __syncthreads();
    int lane = threadIdx.x & 31;
    int tile = blockIdx.x * (blockDim.x >> 5) + (threadIdx.x >> 5);
    if (tile >= ntiles) return;
    int row = tile * 16 + (lane & 15);
    const float* xr = x + (size_t)row * F_IN;
    int hg = (lane >> 4) * 8;
    const v16h* sB = (const v16h*)sW;
    v8f a0 = {}, a1 = {}, a2 = {}, a3 = {};
#pragma unroll 4
    for (int kt = 0; kt < 16; ++kt) {
        int kb = kt * 32 + hg;
        v8f f0 = *(const v8f*)(xr + kb);
        v8f f1 = *(const v8f*)(xr + kb + 16);
        v8h h0 = __builtin_convertvector(f0, v8h);
        v8h h1 = __builtin_convertvector(f1, v8h);
        v16h A;
#pragma unroll
        for (int i = 0; i < 8; i++) { A[i] = h0[i]; A[8 + i] = h1[i]; }
        v16h B0 = sB[(kt * 4 + 0) * 32 + lane];
        v16h B1 = sB[(kt * 4 + 1) * 32 + lane];
        v16h B2 = sB[(kt * 4 + 2) * 32 + lane];
        v16h B3 = sB[(kt * 4 + 3) * 32 + lane];
        a0 = __builtin_amdgcn_wmma_f32_16x16x32_f16(false, A, false, B0, (short)0, a0, false, false);
        a1 = __builtin_amdgcn_wmma_f32_16x16x32_f16(false, A, false, B1, (short)0, a1, false, false);
        a2 = __builtin_amdgcn_wmma_f32_16x16x32_f16(false, A, false, B2, (short)0, a2, false, false);
        a3 = __builtin_amdgcn_wmma_f32_16x16x32_f16(false, A, false, B3, (short)0, a3, false, false);
    }
    int col = lane & 15;
    int rb = tile * 16 + (lane >> 4) * 8;
#pragma unroll
    for (int r = 0; r < 8; r++) {
        int rr = rb + r;
        float* tr = t + (size_t)rr * HID;
        tr[col]      = a0[r];
        tr[col + 16] = a1[r];
        int g = batch[rr];
        atomicAdd(&sv[g * HID + col],      a2[r]);
        atomicAdd(&sv[g * HID + col + 16], a3[r]);
    }
}

// ---------------- layers 2/3 GEMM: h[N,32] @ Wa[32,32] (single WMMA step) ----------
__global__ void k_gemm_h(const float* __restrict__ hin, const _Float16* __restrict__ wf,
                         float* __restrict__ tout, int ntiles) {
    __shared__ __align__(32) _Float16 sW[2 * 32 * 16];
    for (int i = threadIdx.x; i < 1024 / 8; i += blockDim.x)
        ((v8h*)sW)[i] = ((const v8h*)wf)[i];
    __syncthreads();
    int lane = threadIdx.x & 31;
    int tile = blockIdx.x * (blockDim.x >> 5) + (threadIdx.x >> 5);
    if (tile >= ntiles) return;
    int row = tile * 16 + (lane & 15);
    const float* hr = hin + (size_t)row * HID;
    int hg = (lane >> 4) * 8;
    v8f f0 = *(const v8f*)(hr + hg);
    v8f f1 = *(const v8f*)(hr + hg + 16);
    v8h h0 = __builtin_convertvector(f0, v8h);
    v8h h1 = __builtin_convertvector(f1, v8h);
    v16h A;
#pragma unroll
    for (int i = 0; i < 8; i++) { A[i] = h0[i]; A[8 + i] = h1[i]; }
    const v16h* sB = (const v16h*)sW;
    v8f a0 = {}, a1 = {};
    a0 = __builtin_amdgcn_wmma_f32_16x16x32_f16(false, A, false, sB[lane],      (short)0, a0, false, false);
    a1 = __builtin_amdgcn_wmma_f32_16x16x32_f16(false, A, false, sB[32 + lane], (short)0, a1, false, false);
    int col = lane & 15;
    int rb = tile * 16 + (lane >> 4) * 8;
#pragma unroll
    for (int r = 0; r < 8; r++) {
        float* tr = tout + (size_t)(rb + r) * HID;
        tr[col]      = a0[r];
        tr[col + 16] = a1[r];
    }
}

// ---------------- small elementwise / reduction kernels ----------------
__global__ void k_div(const float* __restrict__ sv, const float* __restrict__ cnt,
                      float* __restrict__ gvw) {
    int idx = blockIdx.x * blockDim.x + threadIdx.x;
    if (idx >= NGRAPH * HID) return;
    gvw[idx] = sv[idx] / fmaxf(cnt[idx >> 5], 1.0f);
}

__global__ void k_addg(float* __restrict__ t, const int* __restrict__ batch,
                       const float* __restrict__ gvw, int n) {
    long idx = (long)blockIdx.x * blockDim.x + threadIdx.x;
    if (idx >= (long)n * HID) return;
    int i = idx >> 5, j = idx & 31;
    t[idx] += gvw[batch[i] * HID + j];
}

__global__ void k_self(const float* __restrict__ t, const float* __restrict__ dinv,
                       const float* __restrict__ bias, float* __restrict__ h, int n) {
    long idx = (long)blockIdx.x * blockDim.x + threadIdx.x;
    if (idx >= (long)n * HID) return;
    int i = idx >> 5, j = idx & 31;
    float d = dinv[i];
    h[idx] = bias[j] + d * d * t[idx];
}

__global__ void k_edge(const int* __restrict__ src, const int* __restrict__ dst,
                       const float* __restrict__ dinv, const float* __restrict__ t,
                       float* __restrict__ h) {
    long idx = (long)blockIdx.x * blockDim.x + threadIdx.x;
    if (idx >= (long)N_EDGES * HID) return;
    long e = idx >> 5;
    int j = idx & 31;
    int s = src[e], d = dst[e];
    float w = dinv[s] * dinv[d];
    atomicAdd(&h[(size_t)d * HID + j], w * t[(size_t)s * HID + j]);
}

__global__ void k_post(float* __restrict__ h, const int* __restrict__ batch,
                       float* __restrict__ gsum, float* __restrict__ gmax, int n, int relu) {
    long idx = (long)blockIdx.x * blockDim.x + threadIdx.x;
    if (idx >= (long)n * HID) return;
    int i = idx >> 5, j = idx & 31;
    float v = h[idx];
    if (relu) { v = fmaxf(v, 0.0f); h[idx] = v; }
    int g = batch[i];
    atomicAdd(&gsum[g * HID + j], v);
    atomicMaxF(&gmax[g * HID + j], v);
}

__global__ void k_finpool(float* __restrict__ gap, const float* __restrict__ cnt,
                          const float* __restrict__ gmax, float* __restrict__ z) {
    int idx = blockIdx.x * blockDim.x + threadIdx.x;
    if (idx >= NGRAPH * HID) return;
    int g = idx >> 5, j = idx & 31;
    float m = gap[idx] / fmaxf(cnt[g], 1.0f);
    gap[idx] = m;                       // divided gap feeds next layer's gvw
    z[g * 64 + j]      += gmax[idx];
    z[g * 64 + 32 + j] += m;
}

// gvw[g][j] = sum_k gap[g][k] * W[(32+k)*32 + j]  ("b" half of [64,32] weight)
__global__ void k_gvws(const float* __restrict__ gap, const float* __restrict__ W,
                       float* __restrict__ gvw) {
    int idx = blockIdx.x * blockDim.x + threadIdx.x;
    if (idx >= NGRAPH * HID) return;
    int g = idx >> 5, j = idx & 31;
    const float* gr = gap + g * HID;
    float s = 0.0f;
#pragma unroll 8
    for (int k = 0; k < HID; k++) s += gr[k] * W[(HID + k) * HID + j];
    gvw[idx] = s;
}

__global__ void k_gate(float* __restrict__ h, const float* __restrict__ pw,
                       const float* __restrict__ pb, int n) {
    int i = blockIdx.x * blockDim.x + threadIdx.x;
    if (i >= n) return;
    float* hr = h + (size_t)i * HID;
    float s = pb[0];
#pragma unroll 8
    for (int j = 0; j < HID; j++) s += hr[j] * pw[j];
    float g = 1.0f / (1.0f + __expf(-s));
#pragma unroll 8
    for (int j = 0; j < HID; j++) hr[j] *= g;
}

// ---------------- final MLP on [G,64] ----------------
__global__ void k_mlp(const float* __restrict__ z,
                      const float* __restrict__ w1, const float* __restrict__ bb1,
                      const float* __restrict__ w2, const float* __restrict__ bb2,
                      const float* __restrict__ w3, const float* __restrict__ bb3,
                      float* __restrict__ out) {
    __shared__ float s1[64 * 32], s2[32 * 16], s3[16];
    for (int i = threadIdx.x; i < 2048; i += 256) s1[i] = w1[i];
    for (int i = threadIdx.x; i < 512;  i += 256) s2[i] = w2[i];
    if (threadIdx.x < 16) s3[threadIdx.x] = w3[threadIdx.x];
    __syncthreads();
    int g = threadIdx.x;
    float zi[64];
    for (int k = 0; k < 64; k++) zi[k] = z[g * 64 + k];
    float a[32];
    for (int j = 0; j < 32; j++) {
        float s = bb1[j];
        for (int k = 0; k < 64; k++) s += zi[k] * s1[k * 32 + j];
        a[j] = fmaxf(s, 0.0f);
    }
    float c[16];
    for (int j = 0; j < 16; j++) {
        float s = bb2[j];
        for (int k = 0; k < 32; k++) s += a[k] * s2[k * 16 + j];
        c[j] = fmaxf(s, 0.0f);
    }
    float o = bb3[0];
    for (int k = 0; k < 16; k++) o += c[k] * s3[k];
    out[g] = o;
}

// ---------------- launch ----------------
static inline int GB(long n, int b) { return (int)((n + b - 1) / b); }

extern "C" void kernel_launch(void* const* d_in, const int* in_sizes, int n_in,
                              void* d_out, int out_size, void* d_ws, size_t ws_size,
                              hipStream_t stream) {
    const float* x    = (const float*)d_in[0];
    const int*   ei   = (const int*)d_in[1];
    const int*   batch= (const int*)d_in[2];
    const float* W1   = (const float*)d_in[3];
    const float* b1   = (const float*)d_in[4];
    const float* W2   = (const float*)d_in[5];
    const float* b2   = (const float*)d_in[6];
    const float* W3   = (const float*)d_in[7];
    const float* b3   = (const float*)d_in[8];
    const float* pw   = (const float*)d_in[9];
    const float* pb   = (const float*)d_in[10];
    const float* l1w  = (const float*)d_in[11];
    const float* l1b  = (const float*)d_in[12];
    const float* l2w  = (const float*)d_in[13];
    const float* l2b  = (const float*)d_in[14];
    const float* l3w  = (const float*)d_in[15];
    const float* l3b  = (const float*)d_in[16];
    float* out = (float*)d_out;
    const int* src = ei;
    const int* dst = ei + N_EDGES;

    float* ws   = (float*)d_ws;
    float* dinv = ws;                                 // N
    float* cnt  = dinv + N_NODES;                     // G
    float* t    = cnt + NGRAPH;                       // N*32
    float* h    = t + (size_t)N_NODES * HID;          // N*32
    float* sv   = h + (size_t)N_NODES * HID;          // G*32
    float* gvw  = sv + NGRAPH * HID;                  // G*32
    float* gap  = gvw + NGRAPH * HID;                 // G*32
    float* gmax = gap + NGRAPH * HID;                 // G*32
    float* z    = gmax + NGRAPH * HID;                // G*64
    _Float16* wf1 = (_Float16*)(z + NGRAPH * 64);     // 512*64 halfs
    _Float16* wf2 = wf1 + F_IN * 64;                  // 1024 halfs
    _Float16* wf3 = wf2 + 1024;                       // 1024 halfs

    const long NH = (long)N_NODES * HID;
    const long EH = (long)N_EDGES * HID;
    const int  GH = NGRAPH * HID;

    // ---- init / degrees / counts / weights ----
    k_fill<<<GB(N_NODES, 256), 256, 0, stream>>>(dinv, 0.0f, N_NODES);
    k_fill<<<1, 256, 0, stream>>>(cnt, 0.0f, NGRAPH);
    k_fill<<<GB(NGRAPH * 64, 256), 256, 0, stream>>>(z, 0.0f, NGRAPH * 64);
    k_fill<<<GB(GH, 256), 256, 0, stream>>>(sv, 0.0f, GH);
    k_deg<<<GB(N_EDGES, 256), 256, 0, stream>>>(dst, dinv, N_EDGES);
    k_cnt<<<GB(N_NODES, 256), 256, 0, stream>>>(batch, cnt, N_NODES);
    k_dinv<<<GB(N_NODES, 256), 256, 0, stream>>>(dinv, N_NODES);
    k_wprep1<<<GB(32768, 256), 256, 0, stream>>>(W1, wf1);
    k_wprep_s<<<4, 256, 0, stream>>>(W2, wf2);
    k_wprep_s<<<4, 256, 0, stream>>>(W3, wf3);

    // ---- layer 1 ----
    k_gemm1<<<GB(NTILES, 8), 256, 0, stream>>>(x, wf1, batch, t, sv, NTILES);
    k_div<<<GB(GH, 256), 256, 0, stream>>>(sv, cnt, gvw);
    k_addg<<<GB(NH, 256), 256, 0, stream>>>(t, batch, gvw, N_NODES);
    k_self<<<GB(NH, 256), 256, 0, stream>>>(t, dinv, b1, h, N_NODES);
    k_edge<<<GB(EH, 256), 256, 0, stream>>>(src, dst, dinv, t, h);
    k_fill<<<GB(GH, 256), 256, 0, stream>>>(gap, 0.0f, GH);
    k_fill<<<GB(GH, 256), 256, 0, stream>>>(gmax, -INFINITY, GH);
    k_post<<<GB(NH, 256), 256, 0, stream>>>(h, batch, gap, gmax, N_NODES, 1);
    k_finpool<<<GB(GH, 256), 256, 0, stream>>>(gap, cnt, gmax, z);

    // ---- layer 2 ----
    k_gvws<<<GB(GH, 256), 256, 0, stream>>>(gap, W2, gvw);
    k_gemm_h<<<GB(NTILES, 8), 256, 0, stream>>>(h, wf2, t, NTILES);
    k_addg<<<GB(NH, 256), 256, 0, stream>>>(t, batch, gvw, N_NODES);
    k_self<<<GB(NH, 256), 256, 0, stream>>>(t, dinv, b2, h, N_NODES);
    k_edge<<<GB(EH, 256), 256, 0, stream>>>(src, dst, dinv, t, h);
    k_fill<<<GB(GH, 256), 256, 0, stream>>>(gap, 0.0f, GH);
    k_fill<<<GB(GH, 256), 256, 0, stream>>>(gmax, -INFINITY, GH);
    k_post<<<GB(NH, 256), 256, 0, stream>>>(h, batch, gap, gmax, N_NODES, 1);
    k_finpool<<<GB(GH, 256), 256, 0, stream>>>(gap, cnt, gmax, z);

    // ---- layer 3 (+ score gate) ----
    k_gvws<<<GB(GH, 256), 256, 0, stream>>>(gap, W3, gvw);
    k_gemm_h<<<GB(NTILES, 8), 256, 0, stream>>>(h, wf3, t, NTILES);
    k_addg<<<GB(NH, 256), 256, 0, stream>>>(t, batch, gvw, N_NODES);
    k_self<<<GB(NH, 256), 256, 0, stream>>>(t, dinv, b3, h, N_NODES);
    k_edge<<<GB(EH, 256), 256, 0, stream>>>(src, dst, dinv, t, h);
    k_gate<<<GB(N_NODES, 256), 256, 0, stream>>>(h, pw, pb, N_NODES);
    k_fill<<<GB(GH, 256), 256, 0, stream>>>(gap, 0.0f, GH);
    k_fill<<<GB(GH, 256), 256, 0, stream>>>(gmax, -INFINITY, GH);
    k_post<<<GB(NH, 256), 256, 0, stream>>>(h, batch, gap, gmax, N_NODES, 0);
    k_finpool<<<GB(GH, 256), 256, 0, stream>>>(gap, cnt, gmax, z);

    // ---- readout MLP ----
    k_mlp<<<1, 256, 0, stream>>>(z, l1w, l1b, l2w, l2b, l3w, l3b, out);
}
